// WanDistAttentionBlock_23639499997744
// MI455X (gfx1250) — compile-verified, matrix operand order checked by reading, not run on hardware
//
#include <hip/hip_runtime.h>
#include <hip/hip_bf16.h>

typedef __bf16 bf16;
typedef __attribute__((ext_vector_type(16))) __bf16 bf16x16;
typedef __attribute__((ext_vector_type(8)))  __bf16 bf16x8;
typedef __attribute__((ext_vector_type(8)))  float  f32x8;
typedef __attribute__((ext_vector_type(4)))  unsigned int u32x4;
typedef __attribute__((ext_vector_type(4)))  int i32x4;
typedef __attribute__((ext_vector_type(8)))  int i32x8;

#define C_DIM 384
#define S_LEN 4096
#define B_DIM 4
#define QKV_DIM 1152
#define N_TOK (B_DIM * S_LEN)           // 16384
#define KBLOCKS (S_LEN / 32)            // 128

// workspace layout (bytes)
#define OFF_XN    0ull                                  // [16384][384] bf16
#define OFF_QKV   (OFF_XN   + (size_t)N_TOK*C_DIM*2)    // [16384][1152] bf16 (V part unused)
#define OFF_VT    (OFF_QKV  + (size_t)N_TOK*QKV_DIM*2)  // [4][384][4096] bf16 (V transposed)
#define OFF_ATT   (OFF_VT   + (size_t)N_TOK*C_DIM*2)    // [16384][384] bf16
#define OFF_WQKV  (OFF_ATT  + (size_t)N_TOK*C_DIM*2)    // [1152][384] bf16
#define OFF_WPROJ (OFF_WQKV + (size_t)QKV_DIM*C_DIM*2)  // [384][384] bf16

#if __has_builtin(__builtin_amdgcn_tensor_load_to_lds)
#define HAS_TDM 1
#else
#define HAS_TDM 0
#endif

__device__ __forceinline__ f32x8 wmma_bf16(bf16x16 a, bf16x16 b, f32x8 c) {
  return __builtin_amdgcn_wmma_f32_16x16x32_bf16(false, a, false, b, (short)0, c, false, false);
}
__device__ __forceinline__ bf16x16 pack16(bf16x8 lo, bf16x8 hi) {
  return __builtin_shufflevector(lo, hi, 0,1,2,3,4,5,6,7,8,9,10,11,12,13,14,15);
}

#if HAS_TDM
// 2D tile DMA: tile_d1 rows of tile_d0 contiguous elements (bf16), row stride = stride0 elems.
__device__ __forceinline__ void tdm_load_2d(unsigned ldsOff, const void* gptr,
                                            unsigned tensor_d0, unsigned tensor_d1,
                                            unsigned tile_d0, unsigned tile_d1,
                                            unsigned stride0) {
  unsigned long long ga = (unsigned long long)(uintptr_t)gptr;
  u32x4 g0;
  g0[0] = 1u;                                   // count=1, user descriptor
  g0[1] = ldsOff;                               // lds_addr
  g0[2] = (unsigned)ga;                         // global_addr[31:0]
  g0[3] = (unsigned)(ga >> 32) | 0x80000000u;   // global_addr[56:32] | type=2
  i32x8 g1;
  g1[0] = 0x00010000;                                            // data_size=1 (2B)
  g1[1] = (int)((tensor_d0 & 0xFFFFu) << 16);                    // tensor_dim0[15:0]
  g1[2] = (int)((tensor_d0 >> 16) | ((tensor_d1 & 0xFFFFu) << 16));
  g1[3] = (int)((tensor_d1 >> 16) | (tile_d0 << 16));            // tile_dim0
  g1[4] = (int)tile_d1;                                          // tile_dim1 (tile_dim2=0)
  g1[5] = (int)stride0;                                          // tensor_dim0_stride[31:0]
  g1[6] = 0; g1[7] = 0;
  i32x4 z4 = {0, 0, 0, 0};
#if __clang_major__ >= 23
  i32x8 z8 = {0, 0, 0, 0, 0, 0, 0, 0};
  __builtin_amdgcn_tensor_load_to_lds(g0, g1, z4, z4, z8, 0);
#else
  __builtin_amdgcn_tensor_load_to_lds(g0, g1, z4, z4, 0);
#endif
}
__device__ __forceinline__ void tdm_wait0() {
#if __has_builtin(__builtin_amdgcn_s_wait_tensorcnt)
  __builtin_amdgcn_s_wait_tensorcnt(0);
#else
  asm volatile("s_wait_tensorcnt 0x0" ::: "memory");
#endif
}
__device__ __forceinline__ void tdm_wait2() {
#if __has_builtin(__builtin_amdgcn_s_wait_tensorcnt)
  __builtin_amdgcn_s_wait_tensorcnt(2);
#else
  asm volatile("s_wait_tensorcnt 0x2" ::: "memory");
#endif
}
#endif

// ---------------- kernel 0: weights f32 -> bf16 ----------------
__global__ void cvt_weights_kernel(const float* __restrict__ wq,
                                   const float* __restrict__ wp,
                                   bf16* __restrict__ wqb, bf16* __restrict__ wpb) {
  int i = blockIdx.x * 256 + threadIdx.x;
  if (i < QKV_DIM * C_DIM) wqb[i] = (bf16)wq[i];
  if (i < C_DIM * C_DIM)   wpb[i] = (bf16)wp[i];
}

// ---------------- kernel 1: RMS norm, write token-major bf16 ----------------
__global__ __launch_bounds__(128) void rmsnorm_kernel(const float* __restrict__ x,
                                                      const float* __restrict__ gamma,
                                                      bf16* __restrict__ xn) {
  int token = blockIdx.x;
  int b = token >> 12;
  int t = token & (S_LEN - 1);
  const float* xp = x + (size_t)b * C_DIM * S_LEN + t;
  __shared__ float red[128];
  int tid = threadIdx.x;
  float v[3]; float ss = 0.f;
#pragma unroll
  for (int i = 0; i < 3; i++) { v[i] = xp[(size_t)(tid + 128 * i) * S_LEN]; ss += v[i] * v[i]; }
  red[tid] = ss; __syncthreads();
#pragma unroll
  for (int s = 64; s > 0; s >>= 1) { if (tid < s) red[tid] += red[tid + s]; __syncthreads(); }
  float l2 = sqrtf(red[0]);
  float scale = 19.595917942265423f / fmaxf(l2, 1e-12f);   // sqrt(384)/max(l2,eps)
  bf16* op = xn + (size_t)token * C_DIM;
#pragma unroll
  for (int i = 0; i < 3; i++) { int ch = tid + 128 * i; op[ch] = (bf16)(v[i] * scale * gamma[ch]); }
}

// ---------------- kernel 2: QKV GEMM; V is emitted transposed ----------------
__global__ __launch_bounds__(128) void qkv_gemm_kernel(const bf16* __restrict__ xn,
                                                       const bf16* __restrict__ wq,
                                                       const float* __restrict__ bq,
                                                       bf16* __restrict__ qkv,
                                                       bf16* __restrict__ vt) {
  int lane = threadIdx.x & 31, wave = threadIdx.x >> 5;
  int wid = blockIdx.x * 4 + wave;
  int mTile = wid / (QKV_DIM / 16), nTile = wid % (QKV_DIM / 16);
  int m0 = mTile * 16, n0 = nTile * 16;
  int lrow = lane & 15, lhi = lane >> 4;
  const bf16* aRow = xn + (size_t)(m0 + lrow) * C_DIM;
  const bf16* bRow = wq + (size_t)(n0 + lrow) * C_DIM;
  f32x8 acc = {};
#pragma unroll
  for (int k0 = 0; k0 < C_DIM; k0 += 32) {
    int kb = k0 + lhi * 8;
    bf16x16 A = pack16(*(const bf16x8*)(aRow + kb), *(const bf16x8*)(aRow + kb + 16));
    bf16x16 B = pack16(*(const bf16x8*)(bRow + kb), *(const bf16x8*)(bRow + kb + 16));
    acc = wmma_bf16(A, B, acc);
  }
  float bias = bq[n0 + lrow];
  if (n0 < 768) {            // Q,K: token-major rows
#pragma unroll
    for (int r = 0; r < 8; r++) {
      int tok = m0 + r + 8 * lhi;
      qkv[(size_t)tok * QKV_DIM + n0 + lrow] = (bf16)(acc[r] + bias);
    }
  } else {                   // V: store transposed vt[b][ch][t]
    int ch = n0 - 768 + lrow;
#pragma unroll
    for (int r = 0; r < 8; r++) {
      int tok = m0 + r + 8 * lhi;
      int b = tok >> 12, t = tok & (S_LEN - 1);
      vt[((size_t)b * C_DIM + ch) * S_LEN + t] = (bf16)(acc[r] + bias);
    }
  }
}

// ------- kernel 3: flash attention (8 waves, double-buffered TDM tiles) -------
__global__ __launch_bounds__(256) void attn_kernel(const bf16* __restrict__ qkv,
                                                   const bf16* __restrict__ vt,
                                                   bf16* __restrict__ attnOut) {
  __shared__ bf16 Ks[2][32][C_DIM];   // K tiles row-major       48KB
  __shared__ bf16 Vts[2][C_DIM][32];  // V^T tiles row-major     48KB
  __shared__ bf16 Ps[8][16][32];      // per-wave P scratch       8KB
  int tid = threadIdx.x, lane = tid & 31, w = tid >> 5;
  int batch = blockIdx.x >> 5;             // 32 blocks per batch (128 queries each)
  int qBlk = blockIdx.x & 31;
  int q0 = qBlk * 128 + w * 16;
  int lrow = lane & 15, lhi = lane >> 4;
  const bf16* qkvB = qkv + (size_t)batch * S_LEN * QKV_DIM;
  const bf16* vtB  = vt + (size_t)batch * C_DIM * S_LEN;
  const bf16* qRow = qkvB + (size_t)(q0 + lrow) * QKV_DIM;     // Q at +0

  f32x8 O[24];
#pragma unroll
  for (int nt = 0; nt < 24; nt++) O[nt] = (f32x8){};
  float m[8], l[8];
#pragma unroll
  for (int r = 0; r < 8; r++) { m[r] = -1e30f; l[r] = 0.f; }
  const float scale = 0.05103103630798288f;   // 1/sqrt(384)

#if HAS_TDM
  if (w == 0) {  // prefetch tiles for key-blocks 0 and 1
#pragma unroll
    for (int p = 0; p < 2; p++) {
      tdm_load_2d((unsigned)(uintptr_t)&Ks[p][0][0],
                  qkvB + (size_t)(p * 32) * QKV_DIM + 384,
                  C_DIM, S_LEN, C_DIM, 32, QKV_DIM);
      tdm_load_2d((unsigned)(uintptr_t)&Vts[p][0][0],
                  vtB + p * 32,
                  S_LEN, C_DIM, 32, C_DIM, S_LEN);
    }
  }
#endif

  for (int kb = 0; kb < KBLOCKS; kb++) {
    int buf = kb & 1;
    int k0 = kb * 32;
#if HAS_TDM
    if (w == 0) {
      if (kb < KBLOCKS - 1) tdm_wait2();   // oldest pair (this tile) done, prefetch in flight
      else                  tdm_wait0();
    }
#else
#pragma unroll
    for (int e = tid; e < 32 * 48; e += 256) {    // K tile: 32 rows x 48 chunks
      int kk = e / 48, g = e % 48;
      *(bf16x8*)&Ks[buf][kk][g * 8] = *(const bf16x8*)(qkvB + (size_t)(k0 + kk) * QKV_DIM + 384 + g * 8);
    }
#pragma unroll
    for (int e = tid; e < C_DIM * 4; e += 256) {  // V^T tile: 384 rows x 4 chunks
      int ch = e / 4, g = e % 4;
      *(bf16x8*)&Vts[buf][ch][g * 8] = *(const bf16x8*)(vtB + (size_t)ch * S_LEN + k0 + g * 8);
    }
#endif
    __syncthreads();                         // tiles[buf] ready

    // scores: 16 queries x 32 keys (A from L2-resident global, B from LDS)
    f32x8 s0 = {}, s1 = {};
    const bf16* k0p = &Ks[buf][lrow][0];
    const bf16* k1p = &Ks[buf][16 + lrow][0];
#pragma unroll
    for (int c0 = 0; c0 < C_DIM; c0 += 32) {
      int kc = c0 + lhi * 8;
      bf16x16 A  = pack16(*(const bf16x8*)(qRow + kc), *(const bf16x8*)(qRow + kc + 16));
      bf16x16 B0 = pack16(*(const bf16x8*)(k0p + kc),  *(const bf16x8*)(k0p + kc + 16));
      bf16x16 B1 = pack16(*(const bf16x8*)(k1p + kc),  *(const bf16x8*)(k1p + kc + 16));
      s0 = wmma_bf16(A, B0, s0);
      s1 = wmma_bf16(A, B1, s1);
    }

    // online softmax (row r lives in VGPR r across a 16-lane group)
#pragma unroll
    for (int r = 0; r < 8; r++) {
      float a = s0[r] * scale, b = s1[r] * scale;
      float mx = fmaxf(a, b);
      mx = fmaxf(mx, __shfl_xor(mx, 1));
      mx = fmaxf(mx, __shfl_xor(mx, 2));
      mx = fmaxf(mx, __shfl_xor(mx, 4));
      mx = fmaxf(mx, __shfl_xor(mx, 8));
      float mn = fmaxf(m[r], mx);
      float p0 = __expf(a - mn), p1 = __expf(b - mn);
      float rs = p0 + p1;
      rs += __shfl_xor(rs, 1);
      rs += __shfl_xor(rs, 2);
      rs += __shfl_xor(rs, 4);
      rs += __shfl_xor(rs, 8);
      float alpha = __expf(m[r] - mn);
      l[r] = l[r] * alpha + rs;
      m[r] = mn;
      Ps[w][r + 8 * lhi][lrow]      = (bf16)p0;
      Ps[w][r + 8 * lhi][16 + lrow] = (bf16)p1;
#pragma unroll
      for (int nt = 0; nt < 24; nt++) O[nt][r] *= alpha;
    }
    __syncthreads();                         // P visible (C-layout -> A-layout round trip)

    bf16x16 PA = pack16(*(const bf16x8*)(&Ps[w][lrow][lhi * 8]),
                        *(const bf16x8*)(&Ps[w][lrow][16 + lhi * 8]));
#pragma unroll
    for (int nt = 0; nt < 24; nt++) {
      const bf16* vrow = &Vts[buf][nt * 16 + lrow][lhi * 8];
      bf16x16 VB = pack16(*(const bf16x8*)(vrow), *(const bf16x8*)(vrow + 16));
      O[nt] = wmma_bf16(PA, VB, O[nt]);
    }

    __syncthreads();                         // all waves done with buf
#if HAS_TDM
    if (w == 0 && kb + 2 < KBLOCKS) {        // prefetch kb+2 into freed buffer
      int kn = k0 + 64;
      tdm_load_2d((unsigned)(uintptr_t)&Ks[buf][0][0],
                  qkvB + (size_t)kn * QKV_DIM + 384,
                  C_DIM, S_LEN, C_DIM, 32, QKV_DIM);
      tdm_load_2d((unsigned)(uintptr_t)&Vts[buf][0][0],
                  vtB + kn,
                  S_LEN, C_DIM, 32, C_DIM, S_LEN);
    }
#endif
  }

  // epilogue: O / l -> attnOut [b][t][c] bf16
  float inv[8];
#pragma unroll
  for (int r = 0; r < 8; r++) inv[r] = 1.0f / l[r];
#pragma unroll
  for (int nt = 0; nt < 24; nt++) {
#pragma unroll
    for (int r = 0; r < 8; r++) {
      int tok = q0 + r + 8 * lhi;
      attnOut[((size_t)batch * S_LEN + tok) * C_DIM + nt * 16 + lrow] = (bf16)(O[nt][r] * inv[r]);
    }
  }
}

// ---------------- kernel 4: proj GEMM + bias + residual ----------------
__global__ __launch_bounds__(128) void proj_gemm_kernel(const bf16* __restrict__ att,
                                                        const bf16* __restrict__ wp,
                                                        const float* __restrict__ bp,
                                                        const float* __restrict__ x,
                                                        float* __restrict__ out) {
  int lane = threadIdx.x & 31, wave = threadIdx.x >> 5;
  int wid = blockIdx.x * 4 + wave;
  int mTile = wid / (C_DIM / 16), nTile = wid % (C_DIM / 16);
  int m0 = mTile * 16, n0 = nTile * 16;
  int lrow = lane & 15, lhi = lane >> 4;
  const bf16* aRow = att + (size_t)(m0 + lrow) * C_DIM;
  const bf16* bRow = wp + (size_t)(n0 + lrow) * C_DIM;
  f32x8 acc = {};
#pragma unroll
  for (int k0 = 0; k0 < C_DIM; k0 += 32) {
    int kb = k0 + lhi * 8;
    bf16x16 A = pack16(*(const bf16x8*)(aRow + kb), *(const bf16x8*)(aRow + kb + 16));
    bf16x16 B = pack16(*(const bf16x8*)(bRow + kb), *(const bf16x8*)(bRow + kb + 16));
    acc = wmma_bf16(A, B, acc);
  }
  float bias = bp[n0 + lrow];
#pragma unroll
  for (int r = 0; r < 8; r++) {
    int tok = m0 + r + 8 * lhi;
    int b = tok >> 12, t = tok & (S_LEN - 1);
    size_t idx = ((size_t)b * C_DIM + n0 + lrow) * S_LEN + t;
    out[idx] = acc[r] + bias + x[idx];
  }
}

extern "C" void kernel_launch(void* const* d_in, const int* in_sizes, int n_in,
                              void* d_out, int out_size, void* d_ws, size_t ws_size,
                              hipStream_t stream) {
  (void)in_sizes; (void)n_in; (void)out_size; (void)ws_size;
  const float* x      = (const float*)d_in[0];
  const float* gamma  = (const float*)d_in[1];
  const float* w_qkv  = (const float*)d_in[2];
  const float* b_qkv  = (const float*)d_in[3];
  const float* w_proj = (const float*)d_in[4];
  const float* b_proj = (const float*)d_in[5];
  float* out = (float*)d_out;
  char* ws = (char*)d_ws;
  bf16* xn   = (bf16*)(ws + OFF_XN);
  bf16* qkv  = (bf16*)(ws + OFF_QKV);
  bf16* vt   = (bf16*)(ws + OFF_VT);
  bf16* att  = (bf16*)(ws + OFF_ATT);
  bf16* wqb  = (bf16*)(ws + OFF_WQKV);
  bf16* wpb  = (bf16*)(ws + OFF_WPROJ);

  cvt_weights_kernel<<<(QKV_DIM * C_DIM + 255) / 256, 256, 0, stream>>>(w_qkv, w_proj, wqb, wpb);
  rmsnorm_kernel<<<N_TOK, 128, 0, stream>>>(x, gamma, xn);
  qkv_gemm_kernel<<<(N_TOK / 16) * (QKV_DIM / 16) / 4, 128, 0, stream>>>(xn, wqb, b_qkv, qkv, vt);
  attn_kernel<<<B_DIM * (S_LEN / 128), 256, 0, stream>>>(qkv, vt, att);
  proj_gemm_kernel<<<(N_TOK / 16) * (C_DIM / 16) / 4, 128, 0, stream>>>(att, wpb, b_proj, x, out);
}
